// GVAE_48567490183711
// MI455X (gfx1250) — matrix operand order repaired and checked by reading, not compiled, not run on hardware
//
#include <hip/hip_runtime.h>
#include <hip/hip_bf16.h>

typedef _Float16 half_t;
typedef __attribute__((ext_vector_type(16))) _Float16 v16h;
typedef __attribute__((ext_vector_type(8)))  _Float16 v8h;
typedef __attribute__((ext_vector_type(8)))  float    v8f;

#define N_NODES 12000
#define N_EDGES 192000
#define F_IN    128
#define F_HID   256
#define F_OUT   64

// ---------------------------------------------------------------- helpers
__global__ void k_init_deg(float* deg, int n) {
  int i = blockIdx.x * blockDim.x + threadIdx.x;
  if (i < n) deg[i] = 1.0f;                      // self-loop
}

__global__ void k_count_deg(float* deg, const int* __restrict__ dst, int e) {
  int i = blockIdx.x * blockDim.x + threadIdx.x;
  if (i < e) atomicAdd(&deg[dst[i]], 1.0f);
}

__global__ void k_dinv(const float* __restrict__ deg, float* __restrict__ dinv, int n) {
  int i = blockIdx.x * blockDim.x + threadIdx.x;
  if (i < n) dinv[i] = rsqrtf(deg[i]);
}

__global__ void k_cvt_f16(const float* __restrict__ in, half_t* __restrict__ out, int n) {
  int i = blockIdx.x * blockDim.x + threadIdx.x;
  if (i < n) out[i] = (half_t)in[i];
}

// w [K x Nn] row-major f32  ->  wt [Nn x K] row-major f16 (transposed)
__global__ void k_cvt_tr_f16(const float* __restrict__ w, half_t* __restrict__ wt,
                             int K, int Nn) {
  int i = blockIdx.x * blockDim.x + threadIdx.x;
  if (i < K * Nn) {
    int k = i / Nn, n = i % Nn;
    wt[(size_t)n * K + k] = (half_t)w[i];
  }
}

// out[i,f] = dinv[i]^2 * xw[i,f]   (self-loop contribution, also zero-inits accumulator)
__global__ void k_self_init(float* __restrict__ out, const float* __restrict__ xw,
                            const float* __restrict__ dinv, int n, int F) {
  int i = blockIdx.x * blockDim.x + threadIdx.x;
  if (i < n * F) {
    int r = i / F;
    float dv = dinv[r];
    out[i] = dv * dv * xw[i];
  }
}

// one block per edge, one thread per feature
__global__ void k_edge_scatter(float* __restrict__ out, const float* __restrict__ xw,
                               const float* __restrict__ dinv,
                               const int* __restrict__ src, const int* __restrict__ dst,
                               int F) {
  int e = blockIdx.x;
  int f = threadIdx.x;
  int s = src[e], d = dst[e];
  float c = dinv[s] * dinv[d];
  atomicAdd(&out[(size_t)d * F + f], c * xw[(size_t)s * F + f]);
}

__global__ void k_bias_relu_cvt(const float* __restrict__ acc, const float* __restrict__ b,
                                half_t* __restrict__ outh, int n, int F) {
  int i = blockIdx.x * blockDim.x + threadIdx.x;
  if (i < n * F) {
    float v = acc[i] + b[i % F];
    v = fmaxf(v, 0.0f);
    outh[i] = (half_t)v;
  }
}

// z (f32, in-place in d_out tail) += bias; also emit f16 copy for decoder GEMM
__global__ void k_bias_z(float* __restrict__ z, const float* __restrict__ b,
                         half_t* __restrict__ zh, int n, int F) {
  int i = blockIdx.x * blockDim.x + threadIdx.x;
  if (i < n * F) {
    float v = z[i] + b[i % F];
    z[i] = v;
    zh[i] = (half_t)v;
  }
}

// ---------------------------------------------------------------- WMMA GEMM
// C[M x Nn] = A[M x K] * Bt[Nn x K]^T  (+bias, +activation)
// 256 threads = 8 waves; wave tile 32x32 (2x2 WMMA 16x16x32 f16), block tile 64x128.
// Requires M % 32 == 0, Nn % 32 == 0, K % 32 == 0 (all shapes here satisfy this).
__launch_bounds__(256)
__global__ void k_wmma_gemm(const half_t* __restrict__ A, const half_t* __restrict__ Bt,
                            const float* __restrict__ bias,
                            float* __restrict__ outF, half_t* __restrict__ outH,
                            int M, int Nn, int K, int act) {
  const int lane = threadIdx.x & 31;
  const int wave = threadIdx.x >> 5;
  const int wm = wave & 1;           // 2 waves along M
  const int wn = wave >> 1;          // 4 waves along N
  const int m0 = blockIdx.x * 64  + wm * 32;
  const int n0 = blockIdx.y * 128 + wn * 32;
  if (m0 >= M || n0 >= Nn) return;   // full-tile skip (dims are multiples of 32)

  const int cl = lane & 15;          // lane within half-wave
  const int hi = lane >> 4;          // half-wave select

  v8f acc[2][2] = {};
  union AU { v16h v; v8h h[2]; };

  for (int k0 = 0; k0 < K; k0 += 32) {
    // A fragments: lane row = m0+mi*16+cl; halves = K k0+hi*16 .. +15
    AU a[2];
#pragma unroll
    for (int mi = 0; mi < 2; ++mi) {
      const half_t* ap = A + (size_t)(m0 + mi * 16 + cl) * K + k0 + hi * 16;
      a[mi].h[0] = *(const v8h*)(ap);
      a[mi].h[1] = *(const v8h*)(ap + 8);
    }
    // B fragments from Bt (row-major Nn x K): lane col = n0+ni*16+cl; halves = K k0+hi*16 .. +15
    v16h b[2];
#pragma unroll
    for (int ni = 0; ni < 2; ++ni) {
      const half_t* bp = Bt + (size_t)(n0 + ni * 16 + cl) * K + k0 + hi * 16;
      b[ni] = *(const v16h*)(bp);
    }
#pragma unroll
    for (int mi = 0; mi < 2; ++mi)
#pragma unroll
      for (int ni = 0; ni < 2; ++ni)
        acc[mi][ni] = __builtin_amdgcn_wmma_f32_16x16x32_f16(
            false, a[mi].v, false, b[ni], (short)0, acc[mi][ni], false, false);
  }

  // epilogue: C/D layout -> lane col = cl, vgpr j row = j + 8*hi
#pragma unroll
  for (int ni = 0; ni < 2; ++ni) {
    int col = n0 + ni * 16 + cl;
    float bv = bias ? bias[col] : 0.0f;
#pragma unroll
    for (int mi = 0; mi < 2; ++mi) {
#pragma unroll
      for (int j = 0; j < 8; ++j) {
        int row = m0 + mi * 16 + hi * 8 + j;
        float v = acc[mi][ni][j] + bv;
        if (act == 1) v = fmaxf(v, 0.0f);
        else if (act == 2) v = 1.0f / (1.0f + __expf(-v));
        size_t o = (size_t)row * Nn + col;
        if (outF) outF[o] = v;
        if (outH) outH[o] = (half_t)v;
      }
    }
  }
}

// ---------------------------------------------------------------- launch
extern "C" void kernel_launch(void* const* d_in, const int* in_sizes, int n_in,
                              void* d_out, int out_size, void* d_ws, size_t ws_size,
                              hipStream_t stream) {
  const float* x   = (const float*)d_in[0];
  const int*   ei  = (const int*)d_in[1];
  const int*   src = ei;
  const int*   dst = ei + N_EDGES;
  const float* w1  = (const float*)d_in[2];
  const float* b1  = (const float*)d_in[3];
  const float* w2  = (const float*)d_in[4];
  const float* b2  = (const float*)d_in[5];
  const float* lw1 = (const float*)d_in[6];
  const float* lb1 = (const float*)d_in[7];
  const float* lw2 = (const float*)d_in[8];
  const float* lb2 = (const float*)d_in[9];

  float*  adj = (float*)d_out;                                   // [N x N]
  float*  z   = (float*)d_out + (size_t)N_NODES * N_NODES;       // [N x OUT]

  // workspace carve-up (256B aligned)
  char* p = (char*)d_ws;
  auto carve = [&](size_t bytes) { char* r = p; p += (bytes + 255) & ~(size_t)255; return r; };
  float*  deg  = (float*) carve((size_t)N_NODES * 4);
  float*  dinv = (float*) carve((size_t)N_NODES * 4);
  half_t* xh   = (half_t*)carve((size_t)N_NODES * F_IN  * 2);
  half_t* w1t  = (half_t*)carve((size_t)F_HID * F_IN  * 2);      // [256 x 128]
  half_t* w2t  = (half_t*)carve((size_t)F_OUT * F_HID * 2);      // [64 x 256]
  half_t* l1t  = (half_t*)carve((size_t)F_HID * F_OUT * 2);      // [256 x 64]
  half_t* l2t  = (half_t*)carve((size_t)F_IN  * F_HID * 2);      // [128 x 256]
  float*  xw   = (float*) carve((size_t)N_NODES * F_HID * 4);
  float*  hacc = (float*) carve((size_t)N_NODES * F_HID * 4);
  half_t* hh   = (half_t*)carve((size_t)N_NODES * F_HID * 2);
  float*  hw   = (float*) carve((size_t)N_NODES * F_OUT * 4);
  half_t* zh   = (half_t*)carve((size_t)N_NODES * F_OUT * 2);
  half_t* th   = (half_t*)carve((size_t)N_NODES * F_HID * 2);
  half_t* dh   = (half_t*)carve((size_t)N_NODES * F_IN  * 2);
  (void)ws_size; (void)in_sizes; (void)n_in; (void)out_size;

  auto cdiv = [](int a, int b) { return (a + b - 1) / b; };

  // degrees / normalization
  k_init_deg <<<cdiv(N_NODES, 256), 256, 0, stream>>>(deg, N_NODES);
  k_count_deg<<<cdiv(N_EDGES, 256), 256, 0, stream>>>(deg, dst, N_EDGES);
  k_dinv     <<<cdiv(N_NODES, 256), 256, 0, stream>>>(deg, dinv, N_NODES);

  // f16 conversions (x) and transposed f16 weights
  k_cvt_f16   <<<cdiv(N_NODES * F_IN, 256), 256, 0, stream>>>(x, xh, N_NODES * F_IN);
  k_cvt_tr_f16<<<cdiv(F_IN  * F_HID, 256), 256, 0, stream>>>(w1,  w1t, F_IN,  F_HID);
  k_cvt_tr_f16<<<cdiv(F_HID * F_OUT, 256), 256, 0, stream>>>(w2,  w2t, F_HID, F_OUT);
  k_cvt_tr_f16<<<cdiv(F_OUT * F_HID, 256), 256, 0, stream>>>(lw1, l1t, F_OUT, F_HID);
  k_cvt_tr_f16<<<cdiv(F_HID * F_IN,  256), 256, 0, stream>>>(lw2, l2t, F_HID, F_IN);

  // --- encoder layer 1: xw = x@w1 ; scatter ; h = relu(.+b1) ---
  {
    dim3 g(cdiv(N_NODES, 64), cdiv(F_HID, 128));
    k_wmma_gemm<<<g, 256, 0, stream>>>(xh, w1t, nullptr, xw, nullptr,
                                       N_NODES, F_HID, F_IN, 0);
  }
  k_self_init   <<<cdiv(N_NODES * F_HID, 256), 256, 0, stream>>>(hacc, xw, dinv, N_NODES, F_HID);
  k_edge_scatter<<<N_EDGES, F_HID, 0, stream>>>(hacc, xw, dinv, src, dst, F_HID);
  k_bias_relu_cvt<<<cdiv(N_NODES * F_HID, 256), 256, 0, stream>>>(hacc, b1, hh, N_NODES, F_HID);

  // --- encoder layer 2: hw = h@w2 ; scatter -> z (in d_out tail) ; +b2 ---
  {
    dim3 g(cdiv(N_NODES, 64), cdiv(F_OUT, 128));
    k_wmma_gemm<<<g, 256, 0, stream>>>(hh, w2t, nullptr, hw, nullptr,
                                       N_NODES, F_OUT, F_HID, 0);
  }
  k_self_init   <<<cdiv(N_NODES * F_OUT, 256), 256, 0, stream>>>(z, hw, dinv, N_NODES, F_OUT);
  k_edge_scatter<<<N_EDGES, F_OUT, 0, stream>>>(z, hw, dinv, src, dst, F_OUT);
  k_bias_z      <<<cdiv(N_NODES * F_OUT, 256), 256, 0, stream>>>(z, b2, zh, N_NODES, F_OUT);

  // --- decoder MLP: t = relu(z@lw1+lb1) ; d = sigmoid(t@lw2+lb2) ---
  {
    dim3 g(cdiv(N_NODES, 64), cdiv(F_HID, 128));
    k_wmma_gemm<<<g, 256, 0, stream>>>(zh, l1t, lb1, nullptr, th,
                                       N_NODES, F_HID, F_OUT, 1);
  }
  {
    dim3 g(cdiv(N_NODES, 64), cdiv(F_IN, 128));
    k_wmma_gemm<<<g, 256, 0, stream>>>(th, l2t, lb2, nullptr, dh,
                                       N_NODES, F_IN, F_HID, 2);
  }

  // --- adjacency: adj = d @ d^T  (Bt == dh itself); store-bandwidth bound ---
  {
    dim3 g(cdiv(N_NODES, 64), cdiv(N_NODES, 128));
    k_wmma_gemm<<<g, 256, 0, stream>>>(dh, dh, nullptr, adj, nullptr,
                                       N_NODES, N_NODES, F_IN, 0);
  }
}
